// RotaryMultiheadAttention_33526514712738
// MI455X (gfx1250) — matrix operand order repaired and checked by reading, not compiled, not run on hardware
//
#include <hip/hip_runtime.h>
#include <hip/hip_bf16.h>
#include <math.h>

// ---------------------------------------------------------------------------
// Rotary multi-head attention, MI455X (gfx1250), wave32 + WMMA bf16
//   + TDM (tensor_load_to_lds) staging in the scores kernel (TENSORcnt)
//   + async global->LDS double-buffered staging in the ctx kernel (ASYNCcnt).
// B=2, S=2048, D=1024, H=16, DK=64.  RoPE base = 4500 (S=2048 path).
// ---------------------------------------------------------------------------

typedef __bf16 bf16_t;
typedef __attribute__((ext_vector_type(16))) __bf16 v16bf;
typedef __attribute__((ext_vector_type(8)))  __bf16 v8bf;
typedef __attribute__((ext_vector_type(8)))  float  v8f;
typedef __attribute__((vector_size(16))) int          v4i;
typedef __attribute__((vector_size(16))) unsigned int u32x4;
typedef __attribute__((vector_size(16))) int          i32x4;
typedef __attribute__((vector_size(32))) int          i32x8;

constexpr int kB  = 2;
constexpr int kS  = 2048;
constexpr int kD  = 1024;
constexpr int kH  = 16;
constexpr int kDK = 64;

// ---------------- CDNA5 feature detection ----------------------------------
#if defined(__gfx1250__)
  #if __has_builtin(__builtin_amdgcn_global_load_async_to_lds_b128)
    #define RMHA_ASYNC_BUILTIN 1
  #else
    #define RMHA_ASYNC_ASM 1
  #endif
  #if __has_builtin(__builtin_amdgcn_tensor_load_to_lds)
    #define RMHA_TDM 1
  #endif
  #if __has_builtin(__builtin_amdgcn_s_wait_asynccnt)
    #define WAIT_ASYNCCNT(n) __builtin_amdgcn_s_wait_asynccnt((unsigned short)(n))
  #else
    #define WAIT_ASYNCCNT(n) asm volatile("s_wait_asynccnt %0" ::"i"(n) : "memory")
  #endif
  #if __has_builtin(__builtin_amdgcn_s_wait_tensorcnt)
    #define WAIT_TENSORCNT(n) __builtin_amdgcn_s_wait_tensorcnt((unsigned short)(n))
  #else
    #define WAIT_TENSORCNT(n) asm volatile("s_wait_tensorcnt %0" ::"i"(n) : "memory")
  #endif
#else
  #define WAIT_ASYNCCNT(n)  ((void)0)
  #define WAIT_TENSORCNT(n) ((void)0)
#endif

// 16B async copy global -> LDS (ASYNCcnt path).  LDS offset = low 32 bits of
// the flat address (ISA: LDS aperture keeps the byte offset in addr[31:0]).
__device__ inline void cp_async16(bf16_t* lds_dst, const bf16_t* gsrc) {
#if defined(RMHA_ASYNC_BUILTIN)
  __builtin_amdgcn_global_load_async_to_lds_b128(
      (__attribute__((address_space(1))) v4i*)(uintptr_t)gsrc,
      (__attribute__((address_space(3))) v4i*)(unsigned int)(uintptr_t)lds_dst,
      0, 0);
#elif defined(RMHA_ASYNC_ASM)
  unsigned int l = (unsigned int)(uintptr_t)lds_dst;
  asm volatile("global_load_async_to_lds_b128 %0, %1, off"
               ::"v"(l), "v"(gsrc)
               : "memory");
#else
  *(v8bf*)lds_dst = *(const v8bf*)gsrc;  // host / fallback: synchronous
#endif
}

// ---------------- WMMA core ----------------
__device__ inline v8f wmma_bf16(v16bf a, v16bf b, v8f c) {
#if defined(__gfx1250__)
  return __builtin_amdgcn_wmma_f32_16x16x32_bf16(
      false, a, false, b, (short)0, c, false, false);
#else
  return c;
#endif
}

// A-fragment (16x32 bf16): VGPR0-3 = K[8h..8h+7], VGPR4-7 = K[16+8h..23+8h].
__device__ inline v16bf frag_a_bf16(const bf16_t* __restrict__ X, int ld,
                                    int row, int k0, int hv) {
  const bf16_t* p = X + (size_t)row * ld + k0 + 8 * hv;
  v8bf c0 = *(const v8bf*)p;
  v8bf c1 = *(const v8bf*)(p + 16);
  v16bf r;
#pragma unroll
  for (int i = 0; i < 8; ++i) { r[i] = c0[i]; r[i + 8] = c1[i]; }
  return r;
}

__device__ inline v16bf frag_a_f32(const float* __restrict__ X, int ld,
                                   int row, int k0, int hv) {
  const float* p = X + (size_t)row * ld + k0 + 8 * hv;
  v8f c0 = *(const v8f*)p;
  v8f c1 = *(const v8f*)(p + 16);
  v16bf r;
#pragma unroll
  for (int i = 0; i < 8; ++i) { r[i] = (__bf16)c0[i]; r[i + 8] = (__bf16)c1[i]; }
  return r;
}

// B-fragment (32x16): lane holds K[8h..8h+7] and K[16+8h..23+8h] of column n.
__device__ inline v16bf frag_b_bf16(const bf16_t* __restrict__ W, int ld,
                                    int col, int k0, int hv) {
  const bf16_t* p = W + (size_t)col * ld + k0 + 8 * hv;
  v8bf c0 = *(const v8bf*)p;
  v8bf c1 = *(const v8bf*)(p + 16);
  v16bf r;
#pragma unroll
  for (int i = 0; i < 8; ++i) { r[i] = c0[i]; r[i + 8] = c1[i]; }
  return r;
}

__device__ inline v16bf frag_b_f32(const float* __restrict__ W, int ld,
                                   int col, int k0, int hv) {
  const float* p = W + (size_t)col * ld + k0 + 8 * hv;
  v8f c0 = *(const v8f*)p;
  v8f c1 = *(const v8f*)(p + 16);
  v16bf r;
#pragma unroll
  for (int i = 0; i < 8; ++i) { r[i] = (__bf16)c0[i]; r[i + 8] = (__bf16)c1[i]; }
  return r;
}

// ---------------- Generic NT GEMM:  C[M,N] = A[M,K] * W[N,K]^T + bias ------
__global__ __launch_bounds__(256) void rmha_gemm_nt_f32(
    const float* __restrict__ A, const float* __restrict__ W,
    const float* __restrict__ bias, float* __restrict__ C,
    int M, int N, int K) {
  const int lane = threadIdx.x & 31;
  const int wave = threadIdx.x >> 5;
  const int wm = wave >> 2, wn = wave & 3;
  const int hv = lane >> 4, l16 = lane & 15;
  const int mb = blockIdx.y * 128 + wm * 64;
  const int nb = blockIdx.x * 128 + wn * 32;

  v8f acc[4][2] = {};
  for (int k0 = 0; k0 < K; k0 += 32) {
    if (k0 + 32 < K) {
      __builtin_prefetch(A + (size_t)(mb + l16) * K + k0 + 32, 0, 0);
      __builtin_prefetch(W + (size_t)(nb + l16) * K + k0 + 32, 0, 0);
    }
    v16bf af[4];
#pragma unroll
    for (int i = 0; i < 4; ++i)
      af[i] = frag_a_f32(A, K, mb + 16 * i + l16, k0, hv);
    v16bf bf[2];
#pragma unroll
    for (int j = 0; j < 2; ++j)
      bf[j] = frag_b_f32(W, K, nb + 16 * j + l16, k0, hv);
#pragma unroll
    for (int i = 0; i < 4; ++i)
#pragma unroll
      for (int j = 0; j < 2; ++j)
        acc[i][j] = wmma_bf16(af[i], bf[j], acc[i][j]);
  }
#pragma unroll
  for (int i = 0; i < 4; ++i)
#pragma unroll
    for (int j = 0; j < 2; ++j) {
      const int m0 = mb + 16 * i + 8 * hv;
      const int n  = nb + 16 * j + l16;
      const float bv = bias ? bias[n] : 0.0f;
#pragma unroll
      for (int r = 0; r < 8; ++r)
        C[(size_t)(m0 + r) * N + n] = acc[i][j][r] + bv;
    }
}

// ---------------- RoPE + pack to bf16 head layout [B,H,S,DK] ---------------
__global__ __launch_bounds__(256) void rmha_rope_pack(
    const float* __restrict__ Xf, bf16_t* __restrict__ Xh) {
  const int t = blockIdx.x * blockDim.x + threadIdx.x;  // 2^21
  const int j = t & 31;
  const int h = (t >> 5) & 15;
  const int s = (t >> 9) & 2047;
  const int b = (t >> 20) & 1;
  const float* src = Xf + (size_t)(b * kS + s) * kD + h * kDK + 2 * j;
  const float xe = src[0], xo = src[1];
  const float theta = expf(-(float)j * 0.52573954f);  // ln(4500)/16
  const float ang = (float)s * theta;
  float sn, cs;
  sincosf(ang, &sn, &cs);
  bf16_t* dst = Xh + ((size_t)(b * kH + h) * kS + s) * kDK + 2 * j;
  dst[0] = (__bf16)(xe * cs - xo * sn);
  dst[1] = (__bf16)(xe * sn + xo * cs);
}

// ---------------- pack V transposed to bf16 [B,H,DK,S] ---------------------
__global__ __launch_bounds__(256) void rmha_pack_vt(
    const float* __restrict__ Vf, bf16_t* __restrict__ Vt) {
  const int t = blockIdx.x * blockDim.x + threadIdx.x;  // 2^22
  const int s = t & 2047;
  const int d = (t >> 11) & 63;
  const int h = (t >> 17) & 15;
  const int b = (t >> 21) & 1;
  const float v = Vf[(size_t)(b * kS + s) * kD + h * kDK + d];
  Vt[((size_t)(b * kH + h) * kDK + d) * kS + s] = (__bf16)v;
}

// ---------------- scores = (Q Kh^T) / 8, per (b,h) -------------------------
// K-tile (128 rows x 64 K = 16KB bf16, contiguous) staged into LDS by the
// Tensor Data Mover (one D# descriptor, wave 0 issues, TENSORcnt).
__global__ __launch_bounds__(256) void rmha_scores(
    const bf16_t* __restrict__ Qh, const bf16_t* __restrict__ Kh,
    float* __restrict__ attn) {
  const int bh = blockIdx.z;
  const bf16_t* Q  = Qh + (size_t)bh * kS * kDK;
  const bf16_t* Km = Kh + (size_t)bh * kS * kDK;
  float* Ao = attn + (size_t)bh * kS * kS;

  const int lane = threadIdx.x & 31;
  const int wave = threadIdx.x >> 5;
  const int wm = wave >> 2, wn = wave & 3;
  const int hv = lane >> 4, l16 = lane & 15;
  const int mb = blockIdx.y * 128 + wm * 64;
  const int nbTile = blockIdx.x * 128;  // K-rows staged for this block

  __shared__ __align__(64) bf16_t ktile[128 * 64];  // 16 KB
  const bf16_t* gtile = Km + (size_t)nbTile * kDK;  // contiguous 16 KB

#if defined(RMHA_TDM)
  if (wave == 0) {
    const unsigned long long ga = (unsigned long long)(uintptr_t)gtile;
    const unsigned int la = (unsigned int)(uintptr_t)&ktile[0];
    u32x4 g0;
    g0[0] = 1u;                                   // count=1, user mode
    g0[1] = la;                                   // lds_addr
    g0[2] = (unsigned int)ga;                     // global_addr[31:0]
    g0[3] = (unsigned int)((ga >> 32) & 0x01FFFFFFu) | (2u << 30);  // +type=2
    i32x8 g1;
    g1[0] = (int)(1u << 16);                      // data_size=1 (2 bytes)
    g1[1] = (int)((unsigned)kDK << 16);           // tensor_dim0 = 64
    g1[2] = (int)(128u << 16);                    // tensor_dim1 = 128
    g1[3] = (int)((unsigned)kDK << 16);           // tile_dim0 = 64
    g1[4] = 128;                                  // tile_dim1 = 128
    g1[5] = kDK;                                  // tensor_dim0_stride = 64
    g1[6] = 0;
    g1[7] = 0;
    i32x4 gz4 = {0, 0, 0, 0};
    i32x8 gz8 = {0, 0, 0, 0, 0, 0, 0, 0};
    __builtin_amdgcn_tensor_load_to_lds(g0, g1, gz4, gz4, gz8, 0);
    WAIT_TENSORCNT(0);
  }
#else
  // cooperative fallback: 8192 bf16 = 1024 x v8bf
  for (int idx = threadIdx.x; idx < 1024; idx += 256)
    ((v8bf*)ktile)[idx] = ((const v8bf*)gtile)[idx];
#endif
  __syncthreads();

  v8f acc[4][2] = {};
#pragma unroll
  for (int k0 = 0; k0 < kDK; k0 += 32) {
    v16bf af[4];
#pragma unroll
    for (int i = 0; i < 4; ++i)
      af[i] = frag_a_bf16(Q, kDK, mb + 16 * i + l16, k0, hv);
    v16bf bf[2];
#pragma unroll
    for (int j = 0; j < 2; ++j)
      bf[j] = frag_b_bf16(ktile, kDK, wn * 32 + 16 * j + l16, k0, hv);
#pragma unroll
    for (int i = 0; i < 4; ++i)
#pragma unroll
      for (int j = 0; j < 2; ++j)
        acc[i][j] = wmma_bf16(af[i], bf[j], acc[i][j]);
  }
#pragma unroll
  for (int i = 0; i < 4; ++i)
#pragma unroll
    for (int j = 0; j < 2; ++j) {
      const int m0 = mb + 16 * i + 8 * hv;
      const int n  = nbTile + wn * 32 + 16 * j + l16;
#pragma unroll
      for (int r = 0; r < 8; ++r)
        Ao[(size_t)(m0 + r) * kS + n] = acc[i][j][r] * 0.125f;
    }
}

// ---------------- row softmax over 2048 cols, one block per row ------------
__global__ __launch_bounds__(256) void rmha_softmax(float* __restrict__ attn) {
  float* p = attn + (size_t)blockIdx.x * kS;
  const int t = threadIdx.x;
  const int lane = t & 31, wave = t >> 5;
  __shared__ float red[8];

  float x[8];
  float mx = -3.402823466e38f;
#pragma unroll
  for (int i = 0; i < 8; ++i) { x[i] = p[t + 256 * i]; mx = fmaxf(mx, x[i]); }
#pragma unroll
  for (int off = 16; off; off >>= 1) mx = fmaxf(mx, __shfl_xor(mx, off, 32));
  if (lane == 0) red[wave] = mx;
  __syncthreads();
  float gm = red[0];
#pragma unroll
  for (int w = 1; w < 8; ++w) gm = fmaxf(gm, red[w]);
  __syncthreads();

  float sum = 0.0f;
#pragma unroll
  for (int i = 0; i < 8; ++i) { x[i] = expf(x[i] - gm); sum += x[i]; }
#pragma unroll
  for (int off = 16; off; off >>= 1) sum += __shfl_xor(sum, off, 32);
  if (lane == 0) red[wave] = sum;
  __syncthreads();
  float gs = 0.0f;
#pragma unroll
  for (int w = 0; w < 8; ++w) gs += red[w];
  const float inv = 1.0f / gs;
#pragma unroll
  for (int i = 0; i < 8; ++i) p[t + 256 * i] = x[i] * inv;
}

// ---------------- ctx = attn @ V  (per head), write f32 [B,S,D] ------------
// V B-tile (64 x 32 bf16 = 4KB per k-step) double-buffered into LDS via
// GLOBAL_LOAD_ASYNC_TO_LDS_B128 (one async op per thread, ASYNCcnt).
__global__ __launch_bounds__(256) void rmha_ctx(
    const float* __restrict__ attn, const bf16_t* __restrict__ Vt,
    float* __restrict__ ctx) {
  const int bh = blockIdx.z;
  const int b = bh >> 4, h = bh & 15;
  const float*  Ap = attn + (size_t)bh * kS * kS;
  const bf16_t* Vp = Vt + (size_t)bh * kDK * kS;  // [DK][S]

  const int lane = threadIdx.x & 31;
  const int wave = threadIdx.x >> 5;
  const int wm = wave >> 1, wn = wave & 1;
  const int hv = lane >> 4, l16 = lane & 15;
  const int mb = blockIdx.x * 128 + wm * 32;
  const int nb = wn * 32;

  __shared__ __align__(64) bf16_t vtile[2][kDK * 32];  // 2 x 4 KB
  // this thread's staging assignment: 16B = 8 K-elements of one V row
  const int srow = threadIdx.x >> 2;        // 0..63  (DK row)
  const int schk = threadIdx.x & 3;         // 0..3   (16B chunk)

  // prologue: stage k0 = 0 into buffer 0
  cp_async16(&vtile[0][srow * 32 + schk * 8],
             Vp + (size_t)srow * kS + 0 + schk * 8);

  v8f acc[2][2] = {};
  for (int k0 = 0; k0 < kS; k0 += 32) {
    const int cur = (k0 >> 5) & 1;
    if (k0 + 32 < kS) {
      cp_async16(&vtile[cur ^ 1][srow * 32 + schk * 8],
                 Vp + (size_t)srow * kS + (k0 + 32) + schk * 8);
      WAIT_ASYNCCNT(1);   // oldest (current buffer) complete
    } else {
      WAIT_ASYNCCNT(0);
    }
    __syncthreads();      // all waves' staging of `cur` visible

    if (k0 + 32 < kS)
      __builtin_prefetch(Ap + (size_t)(mb + l16) * kS + k0 + 32, 0, 0);
    v16bf af[2];
#pragma unroll
    for (int i = 0; i < 2; ++i)
      af[i] = frag_a_f32(Ap, kS, mb + 16 * i + l16, k0, hv);
    v16bf bf[2];
#pragma unroll
    for (int j = 0; j < 2; ++j)
      bf[j] = frag_b_bf16(vtile[cur], 32, nb + 16 * j + l16, 0, hv);
#pragma unroll
    for (int i = 0; i < 2; ++i)
#pragma unroll
      for (int j = 0; j < 2; ++j)
        acc[i][j] = wmma_bf16(af[i], bf[j], acc[i][j]);

    __syncthreads();      // done reading `cur` before it is re-staged
  }
#pragma unroll
  for (int i = 0; i < 2; ++i)
#pragma unroll
    for (int j = 0; j < 2; ++j) {
      const int m0 = mb + 16 * i + 8 * hv;
      const int n  = nb + 16 * j + l16;
#pragma unroll
      for (int r = 0; r < 8; ++r)
        ctx[(size_t)(b * kS + m0 + r) * kD + h * kDK + n] = acc[i][j][r];
    }
}

// ---------------------------------------------------------------------------
extern "C" void kernel_launch(void* const* d_in, const int* in_sizes, int n_in,
                              void* d_out, int out_size, void* d_ws,
                              size_t ws_size, hipStream_t stream) {
  (void)in_sizes; (void)n_in; (void)out_size; (void)ws_size;
  const float* query = (const float*)d_in[0];
  const float* key   = (const float*)d_in[1];
  const float* value = (const float*)d_in[2];
  const float* Wq = (const float*)d_in[3];
  const float* bq = (const float*)d_in[4];
  const float* Wk = (const float*)d_in[5];
  const float* bk = (const float*)d_in[6];
  const float* Wv = (const float*)d_in[7];
  const float* bv = (const float*)d_in[8];
  const float* Wo = (const float*)d_in[9];
  const float* bo = (const float*)d_in[10];

  float* out  = (float*)d_out;                       // [B,S,D]
  float* attn = out + (size_t)kB * kS * kD;          // [B,H,S,S]

  char* ws = (char*)d_ws;
  float*  Qf  = (float*)(ws + (size_t)0);            // 16 MiB f32 [B*S, D]
  float*  Kf  = (float*)(ws + ((size_t)16 << 20));   // 16 MiB
  float*  Vf  = (float*)(ws + ((size_t)32 << 20));   // 16 MiB
  bf16_t* Qh  = (bf16_t*)(ws + ((size_t)48 << 20));  //  8 MiB [B,H,S,DK]
  bf16_t* Kh  = (bf16_t*)(ws + ((size_t)56 << 20));  //  8 MiB
  bf16_t* Vt  = (bf16_t*)(ws + ((size_t)64 << 20));  //  8 MiB [B,H,DK,S]
  float*  Ctx = (float*)(ws + ((size_t)72 << 20));   // 16 MiB f32 [B*S, D]

  const int M = kB * kS;  // 4096

  dim3 gGemm(kD / 128, M / 128);
  rmha_gemm_nt_f32<<<gGemm, 256, 0, stream>>>(query, Wq, bq, Qf, M, kD, kD);
  rmha_gemm_nt_f32<<<gGemm, 256, 0, stream>>>(key,   Wk, bk, Kf, M, kD, kD);
  rmha_gemm_nt_f32<<<gGemm, 256, 0, stream>>>(value, Wv, bv, Vf, M, kD, kD);

  const int ropeThreads = kB * kS * kH * (kDK / 2);  // 2^21
  rmha_rope_pack<<<ropeThreads / 256, 256, 0, stream>>>(Qf, Qh);
  rmha_rope_pack<<<ropeThreads / 256, 256, 0, stream>>>(Kf, Kh);
  const int packThreads = kB * kH * kDK * kS;        // 2^22
  rmha_pack_vt<<<packThreads / 256, 256, 0, stream>>>(Vf, Vt);

  dim3 gScore(kS / 128, kS / 128, kB * kH);
  rmha_scores<<<gScore, 256, 0, stream>>>(Qh, Kh, attn);

  rmha_softmax<<<kB * kH * kS, 256, 0, stream>>>(attn);

  dim3 gCtx(kS / 128, 1, kB * kH);
  rmha_ctx<<<gCtx, 256, 0, stream>>>(attn, Vt, Ctx);

  rmha_gemm_nt_f32<<<gGemm, 256, 0, stream>>>(Ctx, Wo, bo, out, M, kD, kD);
}
